// MambaMIL_53257594470704
// MI455X (gfx1250) — matrix operand order repaired and checked by reading, not compiled, not run on hardware
//
#include <hip/hip_runtime.h>
#include <hip/hip_bf16.h>

#define LSEQ 16384
#define DM   128
#define DI   256
#define DSN  16
#define DTR  8
#define RATE 10

typedef __attribute__((ext_vector_type(16))) _Float16 v16h;
typedef __attribute__((ext_vector_type(8)))  float    v8f;

// inverse of perm = concat(arange(i, L, 10) for i in 0..9); L=16384 -> groups 0..3 have 1639, 4..9 have 1638
__device__ __forceinline__ int inv_perm(int t) {
  int g = t % RATE;
  return g * 1638 + (g < 4 ? g : 4) + t / RATE;
}

// ---------------- vectorized f32 -> f16 (n multiple of 4) ----------------
__global__ __launch_bounds__(256) void k_f2h4(const float* __restrict__ in, _Float16* __restrict__ out, int n4) {
  int i = blockIdx.x * 256 + threadIdx.x;
  if (i >= n4) return;
  float4 v = ((const float4*)in)[i];
  _Float16 o[4] = {(_Float16)v.x, (_Float16)v.y, (_Float16)v.z, (_Float16)v.w};
  ((float2*)out)[i] = *(float2*)o;
}

// ---------------- pack weight (Nout x K f32, row-major) into WMMA B-fragment tiles ----------------
// out[((nt*KT + kt)*32 + lane)*16 + s] = W[(nt*16+lane%16)*K + kt*32 + (lane/16)*8 + (s&7) + (s>>3)*16], zero for padded rows.
// In the GEMM, lane L then reads its 16 B-halves contiguously (two 16-B loads).
__global__ __launch_bounds__(256) void k_pack_w(const float* __restrict__ W, _Float16* __restrict__ out,
                                                int K, int Nout, int Npad) {
  int i = blockIdx.x * 256 + threadIdx.x;   // over Npad*K
  if (i >= Npad * K) return;
  int n    = i & 15;
  int lane = (i >> 4) & 31;
  int t    = i >> 9;              // nt*KT + kt
  int KT   = K >> 5;
  int kt   = t % KT;
  int nt   = t / KT;
  int row  = nt * 16 + (lane & 15);
  int col  = kt * 32 + (lane >> 4) * 8 + (n & 7) + ((n >> 3) << 4);
  float v = (row < Nout) ? W[(size_t)row * K + col] : 0.f;
  out[i] = (_Float16)v;
}

// ---------------- WMMA GEMM: C[MxNout] = act(A16[MxK] @ W^T + bias) ----------------
// Wpk is the pre-packed B-fragment weight (Npad x K, tile-major as above).
// One wave per 16x16 output tile; 8 waves per block. ACT: 0=none, 1=relu, 2=tanh.
template <int ACT>
__global__ __launch_bounds__(256) void k_wmma_gemm(
    const _Float16* __restrict__ A16, const _Float16* __restrict__ Wpk,
    const float* __restrict__ bias, float* __restrict__ C,
    int M, int K, int Nout, int Npad) {
  const int lane = threadIdx.x & 31;
  const int wv   = threadIdx.x >> 5;
  const int ntiles = Npad >> 4;
  const int KT     = K >> 5;
  const int tile = blockIdx.x * 8 + wv;
  const int mt = tile / ntiles;
  const int nt = tile - mt * ntiles;
  if (mt * 16 >= M) return;              // wave-uniform: EXEC stays all-ones for WMMA
  const int hi  = lane >> 4;             // lane half
  const int l16 = lane & 15;

  const _Float16* arow = A16 + (size_t)(mt * 16 + l16) * K + hi * 8;
  const _Float16* wrow = Wpk + (size_t)nt * KT * 512 + lane * 16;

  v8f acc = {};
  for (int k0 = 0; k0 < K; k0 += 32) {
    // A fragment (16-bit A 16x32 layout): halves 0..7 -> K=k0+hi*8+0..7 ; halves 8..15 -> +16
    v16h a;
    const _Float16* ap = arow + k0;
    ((float4*)&a)[0] = *(const float4*)(ap);
    ((float4*)&a)[1] = *(const float4*)(ap + 16);
    // B fragment: pre-packed, lane's 16 halves contiguous
    v16h b;
    const _Float16* wp = wrow + (size_t)(k0 >> 5) * 512;
    ((float4*)&b)[0] = *(const float4*)(wp);
    ((float4*)&b)[1] = *(const float4*)(wp + 8);

    acc = __builtin_amdgcn_wmma_f32_16x16x32_f16(false, a, false, b, (short)0, acc, false, false);
  }

  const int col = nt * 16 + l16;
  if (col < Nout) {
    float bv = bias ? bias[col] : 0.f;
#pragma unroll
    for (int r = 0; r < 8; ++r) {
      float v = acc[r] + bv;
      if (ACT == 1) v = fmaxf(v, 0.f);
      if (ACT == 2) v = tanhf(v);
      C[(size_t)(mt * 16 + r + hi * 8) * Nout + col] = v;
    }
  }
}

// ---------------- layernorm (wave per token, 4 elems/lane), optional permuted f16 output ----------------
__global__ __launch_bounds__(256) void k_layernorm(
    const float* __restrict__ h, const float* __restrict__ w, const float* __restrict__ b,
    _Float16* __restrict__ out16, float* __restrict__ out32, int permute) {
  const int lane = threadIdx.x & 31;
  const int tok  = blockIdx.x * 8 + (threadIdx.x >> 5);
  const float* hp = h + (size_t)tok * DM;
  float4 v = ((const float4*)hp)[lane];
  float s = v.x + v.y + v.z + v.w;
#pragma unroll
  for (int o = 16; o; o >>= 1) s += __shfl_xor(s, o, 32);
  float mean = s * (1.f / DM);
  float d0 = v.x - mean, d1 = v.y - mean, d2 = v.z - mean, d3 = v.w - mean;
  float vs = d0 * d0 + d1 * d1 + d2 * d2 + d3 * d3;
#pragma unroll
  for (int o = 16; o; o >>= 1) vs += __shfl_xor(vs, o, 32);
  float rstd = rsqrtf(vs * (1.f / DM) + 1e-5f);
  int pos = permute ? inv_perm(tok) : tok;
  int d = lane * 4;
  float o0 = d0 * rstd * w[d + 0] + b[d + 0];
  float o1 = d1 * rstd * w[d + 1] + b[d + 1];
  float o2 = d2 * rstd * w[d + 2] + b[d + 2];
  float o3 = d3 * rstd * w[d + 3] + b[d + 3];
  _Float16 ob[4] = {(_Float16)o0, (_Float16)o1, (_Float16)o2, (_Float16)o3};
  *(float2*)(out16 + (size_t)pos * DM + d) = *(float2*)ob;
  if (out32) {
    float4 of; of.x = o0; of.y = o1; of.z = o2; of.w = o3;
    ((float4*)(out32 + (size_t)pos * DM))[lane] = of;
  }
}

// ---------------- depthwise causal conv4 + SiLU ----------------
__global__ __launch_bounds__(256) void k_conv_silu(
    const float* __restrict__ xz, const float* __restrict__ cw, const float* __restrict__ cb,
    float* __restrict__ xact, _Float16* __restrict__ x16) {
  int idx = blockIdx.x * 256 + threadIdx.x;   // t*DI + d over LSEQ*DI
  int t = idx >> 8;
  int d = idx & 255;
  const float* wr = cw + d * 4;
  float acc = cb[d] + wr[3] * xz[(size_t)t * 512 + d];
  if (t >= 1) acc += wr[2] * xz[(size_t)(t - 1) * 512 + d];
  if (t >= 2) acc += wr[1] * xz[(size_t)(t - 2) * 512 + d];
  if (t >= 3) acc += wr[0] * xz[(size_t)(t - 3) * 512 + d];
  float sv = acc / (1.f + __expf(-acc));
  xact[idx] = sv;
  x16[idx]  = (_Float16)sv;
}

// ---------------- dt = softplus(dbl[:, :8] @ dtw^T + dtb) ----------------
__global__ __launch_bounds__(256) void k_dt(
    const float* __restrict__ dbl, const float* __restrict__ dtw, const float* __restrict__ dtb,
    float* __restrict__ dt) {
  int idx = blockIdx.x * 256 + threadIdx.x;
  int t = idx >> 8;
  int d = idx & 255;
  const float* dr = dbl + (size_t)t * 40;
  const float* wr = dtw + d * 8;
  float s = dtb[d];
#pragma unroll
  for (int r = 0; r < 8; ++r) s += dr[r] * wr[r];
  dt[idx] = (s > 20.f) ? s : log1pf(__expf(s));
}

// ---------------- selective scan: one thread per channel, 16-state recurrence in regs ----------------
// fuses y = (scan + u*D) * silu(z) and emits the f16 A-operand for out_proj
__global__ __launch_bounds__(256) void k_scan(
    const float* __restrict__ dbl, const float* __restrict__ dt,
    const float* __restrict__ xact, const float* __restrict__ xz,
    const float* __restrict__ A_log, const float* __restrict__ Dp,
    _Float16* __restrict__ y16) {
  __shared__ float sBC[32];
  const int d = threadIdx.x;   // 0..255
  float As[DSN], hs[DSN];
#pragma unroll
  for (int s = 0; s < DSN; ++s) { As[s] = -__expf(A_log[d * DSN + s]); hs[s] = 0.f; }
  const float Dv = Dp[d];
  for (int t = 0; t < LSEQ; ++t) {
    __syncthreads();
    if (d < 32) sBC[d] = dbl[(size_t)t * 40 + 8 + d];   // B[0..15], C[0..15]
    __syncthreads();
    if (t + 1 < LSEQ) {   // hide next-step latency behind this step's exp chain
      __builtin_prefetch(dt + (size_t)(t + 1) * DI + d, 0, 0);
      __builtin_prefetch(xact + (size_t)(t + 1) * DI + d, 0, 0);
      __builtin_prefetch(xz + (size_t)(t + 1) * 512 + 256 + d, 0, 0);
    }
    float dtv = dt[(size_t)t * DI + d];
    float u   = xact[(size_t)t * DI + d];
    float du  = dtv * u;
    float y = 0.f;
#pragma unroll
    for (int s = 0; s < DSN; ++s) {
      float dA = __expf(dtv * As[s]);
      hs[s] = hs[s] * dA + du * sBC[s];
      y += hs[s] * sBC[16 + s];
    }
    float z = xz[(size_t)t * 512 + 256 + d];
    float g = z / (1.f + __expf(-z));
    y16[(size_t)t * DI + d] = (_Float16)((y + u * Dv) * g);
  }
}

// ---------------- residual add with inverse permutation ----------------
__global__ __launch_bounds__(256) void k_residual(const float* __restrict__ outp, float* __restrict__ h) {
  int i = blockIdx.x * 256 + threadIdx.x;   // over LSEQ*DM
  int t = i >> 7;
  int d = i & 127;
  h[i] += outp[(size_t)inv_perm(t) * DM + d];
}

// ---------------- attention scalar score per token ----------------
__global__ __launch_bounds__(256) void k_attn_score(
    const float* __restrict__ att, const float* __restrict__ w2, const float* __restrict__ b2,
    float* __restrict__ Araw) {
  int t = blockIdx.x * 256 + threadIdx.x;
  if (t >= LSEQ) return;
  const float* ar = att + (size_t)t * 64;
  float s = b2[0];
#pragma unroll
  for (int j = 0; j < 64; ++j) s += ar[j] * w2[j];
  Araw[t] = s;
}

// ---------------- softmax over L (single block) ----------------
__global__ __launch_bounds__(1024) void k_softmax(const float* __restrict__ Araw, float* __restrict__ Aw) {
  __shared__ float red[32];
  const int tid = threadIdx.x;
  float m = -3.4028235e38f;
  for (int t = tid; t < LSEQ; t += 1024) m = fmaxf(m, Araw[t]);
#pragma unroll
  for (int o = 16; o; o >>= 1) m = fmaxf(m, __shfl_xor(m, o, 32));
  if ((tid & 31) == 0) red[tid >> 5] = m;
  __syncthreads();
  if (tid < 32) {
    float v = red[tid];
#pragma unroll
    for (int o = 16; o; o >>= 1) v = fmaxf(v, __shfl_xor(v, o, 32));
    if (tid == 0) red[0] = v;
  }
  __syncthreads();
  m = red[0];
  __syncthreads();
  float s = 0.f;
  for (int t = tid; t < LSEQ; t += 1024) s += __expf(Araw[t] - m);
#pragma unroll
  for (int o = 16; o; o >>= 1) s += __shfl_xor(s, o, 32);
  if ((tid & 31) == 0) red[tid >> 5] = s;
  __syncthreads();
  if (tid < 32) {
    float v = red[tid];
#pragma unroll
    for (int o = 16; o; o >>= 1) v += __shfl_xor(v, o, 32);
    if (tid == 0) red[0] = v;
  }
  __syncthreads();
  float inv = 1.f / red[0];
  for (int t = tid; t < LSEQ; t += 1024) Aw[t] = __expf(Araw[t] - m) * inv;
}

// ---------------- attention pooling: pooled[d] = sum_t Aw[t]*hnf[t,d] ----------------
__global__ __launch_bounds__(256) void k_pooled(const float* __restrict__ Aw, const float* __restrict__ hnf,
                                                float* __restrict__ pooled) {
  const int d = blockIdx.x;
  float s = 0.f;
  for (int t = threadIdx.x; t < LSEQ; t += 256) s += Aw[t] * hnf[(size_t)t * DM + d];
#pragma unroll
  for (int o = 16; o; o >>= 1) s += __shfl_xor(s, o, 32);
  __shared__ float red[8];
  if ((threadIdx.x & 31) == 0) red[threadIdx.x >> 5] = s;
  __syncthreads();
  if (threadIdx.x == 0) {
    float v = 0.f;
    for (int i = 0; i < 8; ++i) v += red[i];
    pooled[d] = v;
  }
}

// ---------------- classifier head: logits, softmax, argmax ----------------
__global__ void k_head(const float* __restrict__ pooled, const float* __restrict__ cw,
                       const float* __restrict__ cb, float* __restrict__ out) {
  if (threadIdx.x != 0) return;
  float lg[4];
  for (int c = 0; c < 4; ++c) {
    float s = cb[c];
    const float* w = cw + c * DM;
    for (int d = 0; d < DM; ++d) s += pooled[d] * w[d];
    lg[c] = s;
    out[c] = s;
  }
  float m = lg[0]; int am = 0;
  for (int c = 1; c < 4; ++c) if (lg[c] > m) { m = lg[c]; am = c; }
  float se = 0.f;
  for (int c = 0; c < 4; ++c) se += __expf(lg[c] - m);
  for (int c = 0; c < 4; ++c) out[4 + c] = __expf(lg[c] - m) / se;
  out[8] = (float)am;
}

extern "C" void kernel_launch(void* const* d_in, const int* in_sizes, int n_in,
                              void* d_out, int out_size, void* d_ws, size_t ws_size,
                              hipStream_t stream) {
  (void)in_sizes; (void)n_in; (void)out_size; (void)ws_size;
  const float* x        = (const float*)d_in[0];
  const float* fc1_w    = (const float*)d_in[1];
  const float* fc1_b    = (const float*)d_in[2];
  const float* ln_w     = (const float*)d_in[3];
  const float* ln_b     = (const float*)d_in[4];
  const float* in_proj  = (const float*)d_in[5];
  const float* conv_w   = (const float*)d_in[6];
  const float* conv_b   = (const float*)d_in[7];
  const float* x_projw  = (const float*)d_in[8];
  const float* dt_w     = (const float*)d_in[9];
  const float* dt_b     = (const float*)d_in[10];
  const float* A_log    = (const float*)d_in[11];
  const float* Dp       = (const float*)d_in[12];
  const float* out_proj = (const float*)d_in[13];
  const float* norm_w   = (const float*)d_in[14];
  const float* norm_b   = (const float*)d_in[15];
  const float* attn_w1  = (const float*)d_in[16];
  const float* attn_b1  = (const float*)d_in[17];
  const float* attn_w2  = (const float*)d_in[18];
  const float* attn_b2  = (const float*)d_in[19];
  const float* clf_w    = (const float*)d_in[20];
  const float* clf_b    = (const float*)d_in[21];
  float* out = (float*)d_out;

  char* base = (char*)d_ws;
  size_t off = 0;
  auto alloc = [&](size_t bytes) -> void* {
    off = (off + 255) & ~(size_t)255;
    void* p = base + off;
    off += bytes;
    return p;
  };

  // region0 serves as f16 input for fc1 (L*1024*2 B) then as xz (L*512*4 B) -- identical sizes
  void* region0 = alloc((size_t)LSEQ * 2048);
  _Float16* x16  = (_Float16*)region0;
  float*    xz   = (float*)region0;
  _Float16* fc1pk = (_Float16*)alloc((size_t)DM * 1024 * 2);
  _Float16* inppk = (_Float16*)alloc((size_t)2 * 2 * DI * DM * 2);
  _Float16* xppk  = (_Float16*)alloc((size_t)2 * 48 * DI * 2);
  _Float16* oppk  = (_Float16*)alloc((size_t)2 * DM * DI * 2);
  _Float16* awpk  = (_Float16*)alloc((size_t)64 * DM * 2);
  float*    hbuf   = (float*)alloc((size_t)LSEQ * DM * 4);
  _Float16* hn16   = (_Float16*)alloc((size_t)LSEQ * DM * 2);
  float*    xact   = (float*)alloc((size_t)LSEQ * DI * 4);
  _Float16* xact16 = (_Float16*)alloc((size_t)LSEQ * DI * 2);
  float*    dbl    = (float*)alloc((size_t)LSEQ * 40 * 4);
  float*    dtbuf  = (float*)alloc((size_t)LSEQ * DI * 4);
  _Float16* y16    = (_Float16*)alloc((size_t)LSEQ * DI * 2);
  float*    outp   = (float*)alloc((size_t)LSEQ * DM * 4);
  // tail aliases: only live after both mamba layers complete
  float*    hnf    = xact;                                                    // L*DM f32
  _Float16* hnf16  = (_Float16*)((char*)xact + (size_t)LSEQ * DM * 4);        // L*DM f16
  float*    att    = (float*)((char*)xact + (size_t)LSEQ * DM * 4 + (size_t)LSEQ * DM * 2); // L*64 f32
  float*    Aw     = (float*)xact16;                                          // L f32
  float*    pooled = (float*)((char*)xact16 + (size_t)LSEQ * 4 + 256);        // 128 f32

  // ---- input conversion + weight packing into WMMA B-fragment layout ----
  k_f2h4<<<LSEQ * 1024 / 4 / 256, 256, 0, stream>>>(x, x16, LSEQ * 1024 / 4);
  k_pack_w<<<(DM * 1024 + 255) / 256, 256, 0, stream>>>(fc1_w, fc1pk, 1024, DM, DM);
  for (int l = 0; l < 2; ++l) {
    k_pack_w<<<(2 * DI * DM + 255) / 256, 256, 0, stream>>>(
        in_proj + (size_t)l * 2 * DI * DM, inppk + (size_t)l * 2 * DI * DM, DM, 2 * DI, 2 * DI);
    k_pack_w<<<(48 * DI + 255) / 256, 256, 0, stream>>>(
        x_projw + (size_t)l * 40 * DI, xppk + (size_t)l * 48 * DI, DI, 40, 48);
    k_pack_w<<<(DM * DI + 255) / 256, 256, 0, stream>>>(
        out_proj + (size_t)l * DM * DI, oppk + (size_t)l * DM * DI, DI, DM, DM);
  }
  k_pack_w<<<(64 * DM + 255) / 256, 256, 0, stream>>>(attn_w1, awpk, DM, 64, 64);

  // ---- fc1: h = relu(x @ fc1_w^T + b) ----
  k_wmma_gemm<1><<<(LSEQ / 16) * (DM / 16) / 8, 256, 0, stream>>>(x16, fc1pk, fc1_b, hbuf, LSEQ, 1024, DM, DM);

  for (int l = 0; l < 2; ++l) {
    k_layernorm<<<LSEQ / 8, 256, 0, stream>>>(hbuf, ln_w + l * DM, ln_b + l * DM, hn16, (float*)nullptr, 1);
    k_wmma_gemm<0><<<(LSEQ / 16) * (2 * DI / 16) / 8, 256, 0, stream>>>(
        hn16, inppk + (size_t)l * 2 * DI * DM, nullptr, xz, LSEQ, DM, 2 * DI, 2 * DI);
    k_conv_silu<<<LSEQ, 256, 0, stream>>>(xz, conv_w + l * DI * 4, conv_b + l * DI, xact, xact16);
    k_wmma_gemm<0><<<(LSEQ / 16) * (48 / 16) / 8, 256, 0, stream>>>(
        xact16, xppk + (size_t)l * 48 * DI, nullptr, dbl, LSEQ, DI, 40, 48);
    k_dt<<<LSEQ, 256, 0, stream>>>(dbl, dt_w + l * DI * DTR, dt_b + l * DI, dtbuf);
    k_scan<<<1, 256, 0, stream>>>(dbl, dtbuf, xact, xz, A_log + l * DI * DSN, Dp + l * DI, y16);
    k_wmma_gemm<0><<<(LSEQ / 16) * (DM / 16) / 8, 256, 0, stream>>>(
        y16, oppk + (size_t)l * DM * DI, nullptr, outp, LSEQ, DI, DM, DM);
    k_residual<<<LSEQ * DM / 256, 256, 0, stream>>>(outp, hbuf);
  }

  // ---- final norm + gated attention pooling + classifier ----
  k_layernorm<<<LSEQ / 8, 256, 0, stream>>>(hbuf, norm_w, norm_b, hnf16, hnf, 0);
  k_wmma_gemm<2><<<(LSEQ / 16) * (64 / 16) / 8, 256, 0, stream>>>(hnf16, awpk, attn_b1, att, LSEQ, DM, 64, 64);
  k_attn_score<<<LSEQ / 256, 256, 0, stream>>>(att, attn_w2, attn_b2, out + 9);   // A_raw -> out[9..]
  k_softmax<<<1, 1024, 0, stream>>>(out + 9, Aw);
  k_pooled<<<DM, 256, 0, stream>>>(Aw, hnf, pooled);
  k_head<<<1, 32, 0, stream>>>(pooled, clf_w, clf_b, out);
}
